// CplxGaussianRBF_69303592288703
// MI455X (gfx1250) — compile-verified, matrix operand order checked by reading, not run on hardware
//
#include <hip/hip_runtime.h>
#include <hip/hip_bf16.h>

typedef __attribute__((ext_vector_type(16))) _Float16 v16h;
typedef __attribute__((ext_vector_type(8)))  float    v8f;

#define NW     64
#define C_CH   32
#define HW_PIX 4096
#define LOG2E  1.44269504088896340736f

// One wave computes a 16-pixel x 64-center RBF tile for one (b,c) and reduces
// over centers with v_wmma_f32_16x16x32_f16 (two K=32 steps). Columns 0/1 of
// the 16x16 accumulator are the real/imag channel-weighted sums.
//
// RBF argument uses the expanded form:
//   c2*|x-mu|^2 = xr*p1 + xi*p2 + |x|^2*c2 + p0
// with p1=-2*c2*mu_r, p2=-2*c2*mu_i, p0=c2*|mu|^2, c2=-log2(e)/(2*stddev),
// so each element costs 3 FMAs + one v_exp_f32.
__global__ __launch_bounds__(256) void CplxGaussianRBF_wmma_kernel(
    const float* __restrict__ x_real, const float* __restrict__ x_imag,
    const float* __restrict__ w_real, const float* __restrict__ w_imag,
    const float* __restrict__ b_real, const float* __restrict__ b_imag,
    const float* __restrict__ mu_real, const float* __restrict__ mu_imag,
    const float* __restrict__ stddev, float* __restrict__ out)
{
    __shared__ float4 s_k[NW];          // {p1, p2, c2, p0} per center -> 1 b128/center
    __shared__ v16h   s_bfrag[2][32];   // pre-packed f16 B fragments per (kb, lane)

    const int tid  = threadIdx.x;
    const int tile = blockIdx.x & 31;   // 32 tiles of 128 pixels cover HW=4096
    const int bc   = blockIdx.x >> 5;   // flattened b*C + c
    const int c    = bc & (C_CH - 1);

    if (tid < NW) {
        // Center constants.
        const float mur = mu_real[tid];
        const float mui = mu_imag[tid];
        const float c2  = -LOG2E / (2.0f * stddev[tid]);
        s_k[tid] = make_float4(-2.0f * c2 * mur,
                               -2.0f * c2 * mui,
                               c2,
                               c2 * (mur * mur + mui * mui));

        // B-matrix fragments, built once per block (64 threads = 2 kb x 32 lanes).
        const int kbi = tid >> 5;
        const int ln  = tid & 31;
        const int mm  = ln & 15;
        const int hh  = ln >> 4;
        v16h frag = {};   // columns >= 2 stay zero
        if (mm < 2) {
            const float* wsrc = (mm == 0) ? w_real : w_imag;
            const int    wb   = c * NW;
            #pragma unroll
            for (int v = 0; v < 8; ++v) {
                // Same K-striping as the 16-bit A layout (B mirrors A, M<->N).
                const int k0 = kbi * 32 + 8 * hh + ((v < 4) ? (2 * v) : (16 + 2 * (v - 4)));
                frag[2 * v]     = (_Float16)wsrc[wb + k0];
                frag[2 * v + 1] = (_Float16)wsrc[wb + k0 + 1];
            }
        }
        s_bfrag[kbi][ln] = frag;
    }
    __syncthreads();

    const int wave = tid >> 5;
    const int lane = tid & 31;
    const int m    = lane & 15;   // A: pixel row; D: output column
    const int hi   = lane >> 4;

    const int    pixA = tile * 128 + wave * 16 + m;
    const size_t base = (size_t)bc * HW_PIX;

    const float xr = x_real[base + pixA];
    const float xi = x_imag[base + pixA];
    const float h2 = __builtin_fmaf(xr, xr, xi * xi);   // |x|^2, once per lane

    v8f acc = {};
    #pragma unroll
    for (int kbi = 0; kbi < 2; ++kbi) {
        const v16h bmat = s_bfrag[kbi][lane];
        v16h a;
        #pragma unroll
        for (int v = 0; v < 8; ++v) {
            // ISA 16-bit A-matrix 16x32 layout: lanes 0-15 hold K={2v,2v+1}
            // (v<4) / K={16+2(v-4),...} (v>=4); lanes 16-31 shifted by 8.
            const int k0 = kbi * 32 + 8 * hi + ((v < 4) ? (2 * v) : (16 + 2 * (v - 4)));
            const float4 kA = s_k[k0];
            const float4 kB = s_k[k0 + 1];
            const float arg0 = __builtin_fmaf(xr, kA.x,
                               __builtin_fmaf(xi, kA.y,
                               __builtin_fmaf(h2, kA.z, kA.w)));
            const float arg1 = __builtin_fmaf(xr, kB.x,
                               __builtin_fmaf(xi, kB.y,
                               __builtin_fmaf(h2, kB.z, kB.w)));
            a[2 * v]     = (_Float16)__builtin_amdgcn_exp2f(arg0);
            a[2 * v + 1] = (_Float16)__builtin_amdgcn_exp2f(arg1);
        }
        // 8 args: (neg_a, A, neg_b, B, c_mod, C, reuse_a, reuse_b)
        acc = __builtin_amdgcn_wmma_f32_16x16x32_f16(
            false, a, false, bmat, (short)0, acc, false, false);
    }

    // D layout: VGPR r -> M = r + 8*hi, N = lane&15. Only columns 0 (real)
    // and 1 (imag) hold meaningful sums.
    if (m < 2) {
        const float bias = (m == 0) ? b_real[c] : b_imag[c];
        #pragma unroll
        for (int r = 0; r < 8; ++r) {
            const int p = tile * 128 + wave * 16 + hi * 8 + r;
            out[(base + p) * 2 + m] = acc[r] + bias;
        }
    }
}

extern "C" void kernel_launch(void* const* d_in, const int* in_sizes, int n_in,
                              void* d_out, int out_size, void* d_ws, size_t ws_size,
                              hipStream_t stream) {
    (void)in_sizes; (void)n_in; (void)d_ws; (void)ws_size; (void)out_size;
    const float* x_real  = (const float*)d_in[0];
    const float* x_imag  = (const float*)d_in[1];
    const float* w_real  = (const float*)d_in[2];
    const float* w_imag  = (const float*)d_in[3];
    const float* b_real  = (const float*)d_in[4];
    const float* b_imag  = (const float*)d_in[5];
    const float* mu_real = (const float*)d_in[6];
    const float* mu_imag = (const float*)d_in[7];
    const float* stddev  = (const float*)d_in[8];
    float* out = (float*)d_out;

    // grid = B*C * (HW/128) = 4*32*32 = 4096 blocks of 256 threads (8 waves).
    dim3 grid(4096), block(256);
    CplxGaussianRBF_wmma_kernel<<<grid, block, 0, stream>>>(
        x_real, x_imag, w_real, w_imag, b_real, b_imag,
        mu_real, mu_imag, stddev, out);
}
